// TernaryLinear_24206435680470
// MI455X (gfx1250) — matrix operand order repaired and checked by reading, not compiled, not run on hardware
//
#include <hip/hip_runtime.h>

typedef __attribute__((ext_vector_type(16))) _Float16 v16h;
typedef __attribute__((ext_vector_type(8)))  _Float16 v8h;
typedef __attribute__((ext_vector_type(4)))  _Float16 v4h;
typedef __attribute__((ext_vector_type(8)))  float    v8f;
typedef __attribute__((ext_vector_type(4)))  int      v4i;

// address-space-qualified views for the async DMA builtin
typedef __attribute__((address_space(1))) v4i v4i_global;
typedef __attribute__((address_space(3))) v4i v4i_lds;

#define MTOT 8192
#define NTOT 16384
#define KTOT 4096
#define BK 32
#define LDT 40                  // padded LDS stride in halves (32 + 8)
#define KTILES (KTOT / BK)

// ---------------------------------------------------------------------------
// CDNA5 async global->LDS helpers (guarded: fall back to plain copies)
// ---------------------------------------------------------------------------
__device__ __forceinline__ void async_copy16(const _Float16* g, _Float16* l) {
#if __has_builtin(__builtin_amdgcn_global_load_async_to_lds_b128)
    __builtin_amdgcn_global_load_async_to_lds_b128(
        (v4i_global*)g, (v4i_lds*)l, 0, 0);
#else
    *reinterpret_cast<v8h*>(l) = *reinterpret_cast<const v8h*>(g);
#endif
}

__device__ __forceinline__ void wait_async0() {
#if __has_builtin(__builtin_amdgcn_s_wait_asynccnt)
    __builtin_amdgcn_s_wait_asynccnt(0);
#else
    asm volatile("s_wait_asynccnt 0x0" ::: "memory");
#endif
}

// ---------------------------------------------------------------------------
// Pre-pass: one-time conversions (removes all conversion VALU from hot loop)
// ---------------------------------------------------------------------------
__global__ __launch_bounds__(256)
void convert_w_f16(const int* __restrict__ wq, _Float16* __restrict__ wh) {
    size_t t = (size_t)blockIdx.x * 256 + threadIdx.x;   // one int4 per thread
    int4 v = reinterpret_cast<const int4*>(wq)[t];
    v4h h;
    h[0] = (_Float16)(short)v.x;   // ternary {-1,0,1}: exact, single v_cvt_f16_i16
    h[1] = (_Float16)(short)v.y;
    h[2] = (_Float16)(short)v.z;
    h[3] = (_Float16)(short)v.w;
    reinterpret_cast<v4h*>(wh)[t] = h;
}

__global__ __launch_bounds__(256)
void convert_x_split(const float* __restrict__ x,
                     _Float16* __restrict__ xh, _Float16* __restrict__ xl) {
    size_t t = (size_t)blockIdx.x * 256 + threadIdx.x;   // one float4 per thread
    float4 v = reinterpret_cast<const float4*>(x)[t];
    float f[4] = {v.x, v.y, v.z, v.w};
    v4h hh, hl;
#pragma unroll
    for (int e = 0; e < 4; ++e) {
        _Float16 h = (_Float16)f[e];
        hh[e] = h;
        hl[e] = (_Float16)(f[e] - (float)h);  // x = hi + lo: ~22 mantissa bits total
    }
    reinterpret_cast<v4h*>(xh)[t] = hh;
    reinterpret_cast<v4h*>(xl)[t] = hl;
}

// ---------------------------------------------------------------------------
// Main GEMM: 256x256 tile, 512 threads (16 wave32s, each 64x64), BK=32.
// Tiles DMA'd global->LDS via async loads; 32 v_wmma per wave per K-step.
// ---------------------------------------------------------------------------
#define BM2 256
#define BN2 256

__global__ __launch_bounds__(512)
void ternary_gemm_f16ws(const _Float16* __restrict__ xh,
                        const _Float16* __restrict__ xl,
                        const _Float16* __restrict__ wh,
                        const float* __restrict__ scale,
                        const float* __restrict__ bias,
                        float* __restrict__ out)
{
    __shared__ _Float16 sXh[2][BM2 * LDT];   // 40 KB
    __shared__ _Float16 sXl[2][BM2 * LDT];   // 40 KB
    __shared__ _Float16 sW [2][BN2 * LDT];   // 40 KB  (total 120 KB < 320 KB/WGP)

    const int tid  = threadIdx.x;
    const int lane = tid & 31;
    const int wave = tid >> 5;
    const int wm   = wave & 3;     // 4 slabs of 64 rows
    const int wn   = wave >> 2;    // 4 slabs of 64 cols

    // L2 supertile swizzle: 32 (M) x 64 (N) tiles; 16x16-tile panels
    // panel working set: 16*4MB (x planes) + 16*2MB (w) = 96 MB < 192 MB L2
    const int flat  = blockIdx.x;          // 2048 blocks
    const int panel = flat >> 8;           // 0..7 : 2 (M) x 4 (N) panels
    const int q     = flat & 255;
    const int bm    = (panel & 1) * 16 + (q & 15);
    const int bn    = (panel >> 1) * 16 + (q >> 4);
    const int m0    = bm * BM2;
    const int n0    = bn * BN2;

    // each plane tile = 256 rows x 32 halves = 1024 x 16B chunks; 2 per thread
    auto issue_tile = [&](int buf, int k0) {
#pragma unroll
        for (int l = 0; l < 2; ++l) {
            int idx = tid + l * 512;          // 0..1023
            int r   = idx >> 2;               // row 0..255
            int c   = idx & 3;                // 16B chunk in row
            async_copy16(xh + (size_t)(m0 + r) * KTOT + k0 + c * 8,
                         &sXh[buf][r * LDT + c * 8]);
            async_copy16(xl + (size_t)(m0 + r) * KTOT + k0 + c * 8,
                         &sXl[buf][r * LDT + c * 8]);
            async_copy16(wh + (size_t)(n0 + r) * KTOT + k0 + c * 8,
                         &sW [buf][r * LDT + c * 8]);
        }
    };

    // A fragment (16x32 f16): lane L<16 -> M=L, k {0..7,16..23}; L>=16 -> {8..15,24..31}
    auto load_a = [&](const _Float16* As, int i) -> v16h {
        int row = wm * 64 + i * 16 + (lane & 15);
        int kc  = (lane >> 4) * 8;
        v8h lo = *reinterpret_cast<const v8h*>(&As[row * LDT + kc]);
        v8h hi = *reinterpret_cast<const v8h*>(&As[row * LDT + kc + 16]);
        v16h a;
#pragma unroll
        for (int t = 0; t < 8; ++t) { a[t] = lo[t]; a[t + 8] = hi[t]; }
        return a;
    };
    // B fragment (32x16 f16): lane n<16 -> col n, K {0..7,16..23}; n>=16 -> {8..15,24..31}
    auto load_b = [&](const _Float16* B, int j) -> v16h {
        int col = wn * 64 + j * 16 + (lane & 15);
        int kc  = (lane >> 4) * 8;
        v8h lo = *reinterpret_cast<const v8h*>(&B[col * LDT + kc]);
        v8h hi = *reinterpret_cast<const v8h*>(&B[col * LDT + kc + 16]);
        v16h b;
#pragma unroll
        for (int t = 0; t < 8; ++t) { b[t] = lo[t]; b[t + 8] = hi[t]; }
        return b;
    };

    v8f acc[4][4] = {};

    issue_tile(0, 0);

    for (int kt = 0; kt < KTILES; ++kt) {
        const int cur = kt & 1;
        const int nxt = cur ^ 1;

        wait_async0();          // tile `cur` fully landed in LDS (in-order ASYNCcnt)
        __syncthreads();        // all waves see it; all done reading buffer `nxt`
        if (kt + 1 < KTILES) issue_tile(nxt, (kt + 1) * BK);   // overlaps compute

        v16h bf[4];
#pragma unroll
        for (int j = 0; j < 4; ++j) bf[j] = load_b(sW[cur], j);
#pragma unroll
        for (int i = 0; i < 4; ++i) {
            v16h ah = load_a(sXh[cur], i);
            v16h al = load_a(sXl[cur], i);
#pragma unroll
            for (int j = 0; j < 4; ++j) {
                acc[i][j] = __builtin_amdgcn_wmma_f32_16x16x32_f16(
                    false, ah, false, bf[j], (short)0, acc[i][j], false, false);
                acc[i][j] = __builtin_amdgcn_wmma_f32_16x16x32_f16(
                    false, al, false, bf[j], (short)0, acc[i][j], false, false);
            }
        }
    }

    // Epilogue: C layout -> VGPR r is M=r (+8 for lanes 16..31), N = lane&15
    const int mBase = m0 + wm * 64;
    const int nBase = n0 + wn * 64;
#pragma unroll
    for (int j = 0; j < 4; ++j) {
        const int n   = nBase + j * 16 + (lane & 15);
        const float sc = scale[n];
        const float bi = bias[n];
#pragma unroll
        for (int i = 0; i < 4; ++i) {
            const int mrow0 = mBase + i * 16 + ((lane >> 4) << 3);
#pragma unroll
            for (int r = 0; r < 8; ++r) {
                out[(size_t)(mrow0 + r) * NTOT + n] = acc[i][j][r] * sc + bi;
            }
        }
    }
}

// ---------------------------------------------------------------------------
// Fallback: fused kernel (used only if workspace is too small)
// ---------------------------------------------------------------------------
#define BM 128
#define BN 128

__global__ __launch_bounds__(256)
void ternary_linear_fused(const float* __restrict__ x,
                          const int*   __restrict__ wq,
                          const float* __restrict__ scale,
                          const float* __restrict__ bias,
                          float* __restrict__ out)
{
    __shared__ _Float16 AsH[2][BM * LDT];
    __shared__ _Float16 AsL[2][BM * LDT];
    __shared__ _Float16 Bs [2][BN * LDT];

    const int tid  = threadIdx.x;
    const int lane = tid & 31;
    const int wave = tid >> 5;
    const int wm   = wave & 1;
    const int wn   = wave >> 1;

    const int flat  = blockIdx.x;
    const int panel = flat >> 10;
    const int q     = flat & 1023;
    const int bm    = (panel >> 2) * 32 + (q & 31);
    const int bn    = (panel & 3) * 32 + (q >> 5);
    const int m0    = bm * BM;
    const int n0    = bn * BN;

    float4 xq[4];
    int4   wv[4];

    auto load_tiles = [&](int k0) {
#pragma unroll
        for (int l = 0; l < 4; ++l) {
            int idx = tid + l * 256;
            int r = idx >> 3, c4 = idx & 7;
            xq[l] = *reinterpret_cast<const float4*>(x  + (size_t)(m0 + r) * KTOT + k0 + c4 * 4);
            wv[l] = *reinterpret_cast<const int4*>  (wq + (size_t)(n0 + r) * KTOT + k0 + c4 * 4);
        }
    };
    auto store_tiles = [&](int buf) {
#pragma unroll
        for (int l = 0; l < 4; ++l) {
            int idx = tid + l * 256;
            int r = idx >> 3, c4 = idx & 7;
            float fx[4] = {xq[l].x, xq[l].y, xq[l].z, xq[l].w};
            int   wi[4] = {wv[l].x, wv[l].y, wv[l].z, wv[l].w};
            v4h hh, hl, wb;
#pragma unroll
            for (int e = 0; e < 4; ++e) {
                _Float16 h = (_Float16)fx[e];
                hh[e] = h;
                hl[e] = (_Float16)(fx[e] - (float)h);
                wb[e] = (_Float16)(short)wi[e];
            }
            *reinterpret_cast<v4h*>(&AsH[buf][r * LDT + c4 * 4]) = hh;
            *reinterpret_cast<v4h*>(&AsL[buf][r * LDT + c4 * 4]) = hl;
            *reinterpret_cast<v4h*>(&Bs [buf][r * LDT + c4 * 4]) = wb;
        }
    };
    auto load_a = [&](const _Float16* As, int i) -> v16h {
        int row = wm * 64 + i * 16 + (lane & 15);
        int kc  = (lane >> 4) * 8;
        v8h lo = *reinterpret_cast<const v8h*>(&As[row * LDT + kc]);
        v8h hi = *reinterpret_cast<const v8h*>(&As[row * LDT + kc + 16]);
        v16h a;
#pragma unroll
        for (int t = 0; t < 8; ++t) { a[t] = lo[t]; a[t + 8] = hi[t]; }
        return a;
    };
    auto load_b = [&](const _Float16* B, int j) -> v16h {
        int col = wn * 32 + j * 16 + (lane & 15);
        int kc  = (lane >> 4) * 8;
        v8h lo = *reinterpret_cast<const v8h*>(&B[col * LDT + kc]);
        v8h hi = *reinterpret_cast<const v8h*>(&B[col * LDT + kc + 16]);
        v16h b;
#pragma unroll
        for (int t = 0; t < 8; ++t) { b[t] = lo[t]; b[t + 8] = hi[t]; }
        return b;
    };

    v8f acc[4][2] = {};
    load_tiles(0);
    store_tiles(0);
    __syncthreads();

    for (int kt = 0; kt < KTILES; ++kt) {
        const int cur = kt & 1;
        const int nxt = cur ^ 1;
        if (kt + 1 < KTILES) load_tiles((kt + 1) * BK);

        v16h bfr[2];
        bfr[0] = load_b(Bs[cur], 0);
        bfr[1] = load_b(Bs[cur], 1);
#pragma unroll
        for (int i = 0; i < 4; ++i) {
            v16h ah = load_a(AsH[cur], i);
            v16h al = load_a(AsL[cur], i);
#pragma unroll
            for (int j = 0; j < 2; ++j) {
                acc[i][j] = __builtin_amdgcn_wmma_f32_16x16x32_f16(
                    false, ah, false, bfr[j], (short)0, acc[i][j], false, false);
                acc[i][j] = __builtin_amdgcn_wmma_f32_16x16x32_f16(
                    false, al, false, bfr[j], (short)0, acc[i][j], false, false);
            }
        }
        if (kt + 1 < KTILES) store_tiles(nxt);
        __syncthreads();
    }

    const int mBase = m0 + wm * 64;
    const int nBase = n0 + wn * 32;
#pragma unroll
    for (int j = 0; j < 2; ++j) {
        const int n   = nBase + j * 16 + (lane & 15);
        const float sc = scale[n];
        const float bi = bias[n];
#pragma unroll
        for (int i = 0; i < 4; ++i) {
            const int mrow0 = mBase + i * 16 + ((lane >> 4) << 3);
#pragma unroll
            for (int r = 0; r < 8; ++r) {
                out[(size_t)(mrow0 + r) * NTOT + n] = acc[i][j][r] * sc + bi;
            }
        }
    }
}

// ---------------------------------------------------------------------------
extern "C" void kernel_launch(void* const* d_in, const int* in_sizes, int n_in,
                              void* d_out, int out_size, void* d_ws, size_t ws_size,
                              hipStream_t stream) {
    const float* x  = (const float*)d_in[0];
    const int*   wq = (const int*)  d_in[1];
    const float* sc = (const float*)d_in[2];
    const float* bi = (const float*)d_in[3];
    float* out = (float*)d_out;

    const size_t xplane = (size_t)MTOT * KTOT;          // halves per x plane
    const size_t wplane = (size_t)NTOT * KTOT;          // halves for weights
    const size_t WS_NEED = (2 * xplane + wplane) * sizeof(_Float16);  // ~268 MB

    if (ws_size >= WS_NEED) {
        _Float16* xh = (_Float16*)d_ws;
        _Float16* xl = xh + xplane;
        _Float16* wh = xl + xplane;
        convert_x_split<<<(int)(xplane / 1024), 256, 0, stream>>>(x, xh, xl);
        convert_w_f16 <<<(int)(wplane / 1024), 256, 0, stream>>>(wq, wh);
        ternary_gemm_f16ws<<<(MTOT / BM2) * (NTOT / BN2), 512, 0, stream>>>(
            xh, xl, wh, sc, bi, out);
    } else {
        ternary_linear_fused<<<(MTOT / BM) * (NTOT / BN), 256, 0, stream>>>(
            x, wq, sc, bi, out);
    }
}
